// FullDiscretizerWLast_34763465294230
// MI455X (gfx1250) — matrix operand order repaired and checked by reading, not compiled, not run on hardware
//
#include <hip/hip_runtime.h>
#include <hip/hip_bf16.h>
#include <math.h>

// ---------------------------------------------------------------------------
// Types for CDNA5 WMMA
// ---------------------------------------------------------------------------
typedef __attribute__((ext_vector_type(16))) __bf16 v16bf;
typedef __attribute__((ext_vector_type(8)))  float  v8f;

union FragBF { unsigned u[8]; v16bf v; };

__device__ inline unsigned short f2bf(float f) {            // RNE f32 -> bf16
    unsigned u = __float_as_uint(f);
    unsigned r = u + 0x7FFFu + ((u >> 16) & 1u);
    return (unsigned short)(r >> 16);
}

// ---------------------------------------------------------------------------
// Problem constants
// ---------------------------------------------------------------------------
#define BATCH   32768
#define NIN     1024
#define DIN     1064           // NUM_INPUT + 4 + 36
#define KP      1088           // DIN padded to multiple of 32
#define HID     2048
#define NACT    37
#define NP2     64             // NACT padded
#define MAXSTEP 20
#define LSTR    40             // LDS row stride in halfwords (80 B, 16B aligned)

// ---------------------------------------------------------------------------
// Async copy helpers (CDNA5: GLOBAL_LOAD_ASYNC_TO_LDS_B128, ASYNCcnt)
// ---------------------------------------------------------------------------
__device__ __forceinline__ void async_b128(unsigned lds_byte_off, const void* gaddr) {
    asm volatile("global_load_async_to_lds_b128 %0, %1, off"
                 :: "v"(lds_byte_off), "v"((unsigned long long)(size_t)gaddr)
                 : "memory");
}
__device__ __forceinline__ void wait_async_le4() {
    asm volatile("s_wait_asynccnt 0x4" ::: "memory");
}
__device__ __forceinline__ void wait_async_le3() {
    asm volatile("s_wait_asynccnt 0x3" ::: "memory");
}
__device__ __forceinline__ void wait_async_0() {
    asm volatile("s_wait_asynccnt 0x0" ::: "memory");
}

// ---------------------------------------------------------------------------
// Weight / input casts to bf16 (weights pre-TRANSPOSED: W1t[n][k], W2t[n][k])
// ---------------------------------------------------------------------------
__global__ void prep_w1t_kernel(const float* __restrict__ W1, unsigned short* __restrict__ W1t) {
    __shared__ float tile[32][33];
    const int k0 = blockIdx.x * 32, n0 = blockIdx.y * 32;
    const int tx = threadIdx.x, ty = threadIdx.y;            // (32, 8)
    #pragma unroll
    for (int j = 0; j < 4; j++) {
        int kk = ty + 8 * j;
        tile[kk][tx] = (k0 + kk < DIN) ? W1[(size_t)(k0 + kk) * HID + n0 + tx] : 0.f;
    }
    __syncthreads();
    #pragma unroll
    for (int j = 0; j < 4; j++) {
        int nn = ty + 8 * j;
        W1t[(size_t)(n0 + nn) * KP + k0 + tx] = f2bf(tile[tx][nn]);
    }
}

__global__ void prep_w2t_kernel(const float* __restrict__ W2, unsigned short* __restrict__ W2t) {
    int e = blockIdx.x * blockDim.x + threadIdx.x;
    if (e >= NP2 * HID) return;
    int n = e >> 11, k = e & (HID - 1);
    W2t[e] = (n < NACT) ? f2bf(W2[k * NACT + n]) : (unsigned short)0;
}

__global__ void conv_fc_kernel(const float* __restrict__ fc, unsigned short* __restrict__ Xb) {
    size_t e = ((size_t)blockIdx.x * blockDim.x + threadIdx.x) * 4;
    if (e >= (size_t)BATCH * NIN) return;
    size_t i = e >> 10;
    size_t j = e & (NIN - 1);
    const float4 v = *(const float4*)(fc + e);
    unsigned short o0 = f2bf(v.x), o1 = f2bf(v.y), o2 = f2bf(v.z), o3 = f2bf(v.w);
    *(uint2*)(Xb + i * KP + j) = make_uint2((unsigned)o0 | ((unsigned)o1 << 16),
                                            (unsigned)o2 | ((unsigned)o3 << 16));
}

// ---------------------------------------------------------------------------
// Discretizer scan (faithful translation of the jax.lax.scan body)
// ---------------------------------------------------------------------------
struct ScanState {
    float ac0, ac1, delta, norm;
    int   rdac[5];
    int   cam_steps;
    bool  dch, committed;
};

__device__ inline void scan_init(ScanState& s) {
    s.ac0 = 0.f; s.ac1 = 0.f; s.delta = 0.0625f; s.norm = 0.f;
    s.rdac[0] = s.rdac[1] = s.rdac[2] = s.rdac[3] = s.rdac[4] = 0;
    s.cam_steps = 0; s.dch = false; s.committed = false;
}

__device__ inline int scan_step(ScanState& s, const int acts[5], float cam0, float cam1) {
    const int RS[5] = {6, 8, 10, 11, 14};
    s.norm += s.committed ? 0.f : 1.f;
    bool commit = ((fabsf(cam0) < 1e-5f && fabsf(cam1) < 1e-5f) || (s.cam_steps >= 6));
    #pragma unroll
    for (int j = 0; j < 5; j++) commit = commit && (s.rdac[j] == acts[j]);
    int  dac = 0;
    bool modified = commit;
    #pragma unroll
    for (int j = 0; j < 5; j++) {
        bool disc = !modified && (s.rdac[j] != acts[j]);
        if (disc) {
            dac = (acts[j] == 0) ? (s.rdac[j] - 1 + RS[j]) : (acts[j] - 1 + RS[j]);
            s.rdac[j] = acts[j];
            modified = true;
        }
    }
    float d2 = s.delta * 2.f;
    // NB: reference compares both cam0 and cam1 against ac[:,0]  (kept as-is)
    bool dmask = !modified && !s.dch &&
                 ((fabsf(cam0 - s.ac0) > d2) || (fabsf(cam1 - s.ac0) > d2));
    if (dmask) {
        dac = 5;
        s.delta = fminf(fmaxf(d2, -0.5f), 0.5f);
        modified = true;
    }
    float sx = (cam0 < s.ac0) ? -1.f : 1.f;
    float sy = (cam1 < s.ac1) ? -1.f : 1.f;
    int quad = (sx < 0.f) ? ((sy < 0.f) ? 1 : 3) : ((sy < 0.f) ? 2 : 4);
    if (!modified) {
        dac = quad;
        s.ac0 += sx * s.delta;
        s.ac1 += sy * s.delta;
        s.delta *= 0.5f;
        s.dch = true;
        s.cam_steps += 1;
    }
    s.committed = s.committed || commit;
    return dac;
}

__global__ void scan_kernel(const float* __restrict__ camera, const float* __restrict__ rand_u,
                            const int* __restrict__ a0, const int* __restrict__ a1,
                            const int* __restrict__ a2, const int* __restrict__ a3,
                            const int* __restrict__ a4,
                            unsigned short* __restrict__ Xb, int* __restrict__ tgt) {
    int i = blockIdx.x * blockDim.x + threadIdx.x;
    if (i >= BATCH) return;
    float cam0 = camera[2 * i], cam1 = camera[2 * i + 1];
    int acts[5] = {a0[i], a1[i], a2[i], a3[i], a4[i]};

    ScanState s; scan_init(s);
    for (int t = 0; t < MAXSTEP; t++) (void)scan_step(s, acts, cam0, cam1);
    int rs = (int)(rand_u[i] * s.norm);
    if (rs > MAXSTEP - 1) rs = MAXSTEP - 1;

    ScanState p; scan_init(p);
    for (int t = 0; t < rs; t++) (void)scan_step(p, acts, cam0, cam1);
    float ac0 = p.ac0, ac1 = p.ac1, delta = p.delta;
    bool  dch = p.dch;
    int   rd[5] = {p.rdac[0], p.rdac[1], p.rdac[2], p.rdac[3], p.rdac[4]};
    int   dac = scan_step(p, acts, cam0, cam1);

    unsigned short* xr = Xb + (size_t)i * KP + NIN;   // sel region (40) + pad (24)
    xr[0] = f2bf(ac0);
    xr[1] = f2bf(ac1);
    #pragma unroll
    for (int j = 2; j < 38; j++) xr[j] = 0;
    const int OH_OFF[5] = {2, 5, 8, 10, 14};
    #pragma unroll
    for (int j = 0; j < 5; j++) xr[OH_OFF[j] + rd[j]] = 0x3F80;   // bf16 1.0
    xr[38] = f2bf(delta);
    xr[39] = dch ? (unsigned short)0x3F80 : (unsigned short)0;
    #pragma unroll
    for (int j = 40; j < 64; j++) xr[j] = 0;          // pad DIN..KP
    tgt[i] = dac;
}

// ---------------------------------------------------------------------------
// GEMM1: X(32768 x 1088 bf16) @ W1t^T + b1, SiLU -> Hb bf16
// 128x128 tile / 256 thr, async double-buffered LDS, v_wmma_f32_16x16x32_bf16
// ---------------------------------------------------------------------------
__global__ __launch_bounds__(256)
void gemm1_kernel(const unsigned short* __restrict__ Xb,
                  const unsigned short* __restrict__ W1t,
                  const float* __restrict__ b1,
                  unsigned short* __restrict__ Hb) {
    __shared__ alignas(16) unsigned short As[2][128 * LSTR];
    __shared__ alignas(16) unsigned short Bs[2][128 * LSTR];

    const int t    = threadIdx.x;
    const int lane = t & 31, wave = t >> 5;
    const int wm = wave >> 2, wn = wave & 3;   // 2x4 wave grid: 64x32 per wave
    const int g  = lane >> 4, lr = lane & 15;
    const int m0 = blockIdx.y * 128, n0 = blockIdx.x * 128;

    const unsigned asBase = (unsigned)(size_t)&As[0][0];
    const unsigned bsBase = (unsigned)(size_t)&Bs[0][0];
    const unsigned bufBytes = 128 * LSTR * 2;

    v8f acc[4][2];
    #pragma unroll
    for (int mt = 0; mt < 4; mt++)
        #pragma unroll
        for (int nt = 0; nt < 2; nt++)
            acc[mt][nt] = (v8f)(0.f);

    auto issue = [&](int buf, int kb) {
        #pragma unroll
        for (int rc = 0; rc < 2; rc++) {
            int cc = t + rc * 256;
            int r = cc >> 2, sgm = cc & 3;                 // row 0..127, 16B segment 0..3
            unsigned loff = (unsigned)(buf * bufBytes + r * (LSTR * 2) + sgm * 16);
            async_b128(asBase + loff, Xb  + (size_t)(m0 + r) * KP + kb + sgm * 8);
            async_b128(bsBase + loff, W1t + (size_t)(n0 + r) * KP + kb + sgm * 8);
        }
    };

    issue(0, 0);
    const int NTILE = KP / 32;                             // 34
    for (int it = 0; it < NTILE; ++it) {
        const int cur = it & 1;
        if (it + 1 < NTILE) { issue(1 - cur, (it + 1) * 32); wait_async_le4(); }
        else                { wait_async_0(); }
        __syncthreads();

        const unsigned* AsU = (const unsigned*)&As[cur][0];
        const unsigned* BsU = (const unsigned*)&Bs[cur][0];
        FragBF a[4], b[2];
        #pragma unroll
        for (int mt = 0; mt < 4; mt++) {
            int m = wm * 64 + mt * 16 + lr;
            #pragma unroll
            for (int i = 0; i < 8; i++) {
                int kk = ((i < 4) ? 0 : 16) + 8 * g + 2 * (i & 3);   // ISA 16-bit A 16x32 layout
                a[mt].u[i] = AsU[m * (LSTR / 2) + (kk >> 1)];
            }
        }
        #pragma unroll
        for (int nt = 0; nt < 2; nt++) {
            int n = wn * 32 + nt * 16 + lr;
            #pragma unroll
            for (int i = 0; i < 8; i++) {
                int kk = ((i < 4) ? 0 : 16) + 8 * g + 2 * (i & 3);   // same k order as A
                b[nt].u[i] = BsU[n * (LSTR / 2) + (kk >> 1)];
            }
        }
        #pragma unroll
        for (int mt = 0; mt < 4; mt++)
            #pragma unroll
            for (int nt = 0; nt < 2; nt++)
                acc[mt][nt] = __builtin_amdgcn_wmma_f32_16x16x32_bf16(
                    false, a[mt].v, false, b[nt].v, (short)0, acc[mt][nt], false, false);
        __syncthreads();
    }

    // ---- epilogue: + b1, SiLU, store bf16 ----
    float bv[2];
    bv[0] = b1[n0 + wn * 32 + 0 * 16 + lr];
    bv[1] = b1[n0 + wn * 32 + 1 * 16 + lr];
    #pragma unroll
    for (int mt = 0; mt < 4; mt++)
        #pragma unroll
        for (int nt = 0; nt < 2; nt++) {
            int col = n0 + wn * 32 + nt * 16 + lr;
            #pragma unroll
            for (int r = 0; r < 8; r++) {
                int row = m0 + wm * 64 + mt * 16 + 8 * g + r;
                float h = acc[mt][nt][r] + bv[nt];
                float v = h / (1.f + expf(-h));     // h * sigmoid(h)
                Hb[(size_t)row * HID + col] = f2bf(v);
            }
        }
}

// ---------------------------------------------------------------------------
// GEMM2: Hb(32768 x 2048 bf16) @ W2t^T + b2 -> logits f32
// 128x64 tile / 256 thr, async double-buffered LDS
// ---------------------------------------------------------------------------
__global__ __launch_bounds__(256)
void gemm2_kernel(const unsigned short* __restrict__ Hb,
                  const unsigned short* __restrict__ W2t,
                  const float* __restrict__ b2,
                  float* __restrict__ logits) {
    __shared__ alignas(16) unsigned short As[2][128 * LSTR];
    __shared__ alignas(16) unsigned short Bs[2][64 * LSTR];

    const int t    = threadIdx.x;
    const int lane = t & 31, wave = t >> 5;
    const int g  = lane >> 4, lr = lane & 15;
    const int m0 = blockIdx.x * 128;

    const unsigned asBase = (unsigned)(size_t)&As[0][0];
    const unsigned bsBase = (unsigned)(size_t)&Bs[0][0];

    v8f acc[4];
    #pragma unroll
    for (int nt = 0; nt < 4; nt++) acc[nt] = (v8f)(0.f);

    auto issue = [&](int buf, int kb) {
        #pragma unroll
        for (int rc = 0; rc < 2; rc++) {
            int cc = t + rc * 256;
            int r = cc >> 2, sgm = cc & 3;
            async_b128(asBase + (unsigned)(buf * 128 * LSTR * 2 + r * (LSTR * 2) + sgm * 16),
                       Hb + (size_t)(m0 + r) * HID + kb + sgm * 8);
        }
        {
            int r = t >> 2, sgm = t & 3;                   // 64 rows of W2t
            async_b128(bsBase + (unsigned)(buf * 64 * LSTR * 2 + r * (LSTR * 2) + sgm * 16),
                       W2t + (size_t)r * HID + kb + sgm * 8);
        }
    };

    issue(0, 0);
    const int NTILE = HID / 32;                            // 64
    for (int it = 0; it < NTILE; ++it) {
        const int cur = it & 1;
        if (it + 1 < NTILE) { issue(1 - cur, (it + 1) * 32); wait_async_le3(); }
        else                { wait_async_0(); }
        __syncthreads();

        const unsigned* AsU = (const unsigned*)&As[cur][0];
        const unsigned* BsU = (const unsigned*)&Bs[cur][0];
        FragBF a, b[4];
        {
            int m = wave * 16 + lr;
            #pragma unroll
            for (int i = 0; i < 8; i++) {
                int kk = ((i < 4) ? 0 : 16) + 8 * g + 2 * (i & 3);
                a.u[i] = AsU[m * (LSTR / 2) + (kk >> 1)];
            }
        }
        #pragma unroll
        for (int nt = 0; nt < 4; nt++) {
            int n = nt * 16 + lr;
            #pragma unroll
            for (int i = 0; i < 8; i++) {
                int kk = ((i < 4) ? 0 : 16) + 8 * g + 2 * (i & 3);
                b[nt].u[i] = BsU[n * (LSTR / 2) + (kk >> 1)];
            }
        }
        #pragma unroll
        for (int nt = 0; nt < 4; nt++)
            acc[nt] = __builtin_amdgcn_wmma_f32_16x16x32_bf16(
                false, a.v, false, b[nt].v, (short)0, acc[nt], false, false);
        __syncthreads();
    }

    #pragma unroll
    for (int nt = 0; nt < 4; nt++) {
        int col = nt * 16 + lr;
        float bb = (col < NACT) ? b2[col] : 0.f;
        #pragma unroll
        for (int r = 0; r < 8; r++) {
            int row = m0 + wave * 16 + 8 * g + r;
            logits[(size_t)row * NP2 + col] = acc[nt][r] + bb;
        }
    }
}

// ---------------------------------------------------------------------------
// Loss: per row logsumexp over 37 logits, NLL gather
// ---------------------------------------------------------------------------
__global__ void loss_kernel(const float* __restrict__ logits, const int* __restrict__ tgt,
                            float* __restrict__ out) {
    int i = blockIdx.x * blockDim.x + threadIdx.x;
    if (i >= BATCH) return;
    const float* l = logits + (size_t)i * NP2;
    float m = -3.4e38f;
    for (int j = 0; j < NACT; j++) m = fmaxf(m, l[j]);
    float s = 0.f;
    for (int j = 0; j < NACT; j++) s += expf(l[j] - m);
    out[i] = m + logf(s) - l[tgt[i]];
}

// ---------------------------------------------------------------------------
// Launch
// ---------------------------------------------------------------------------
extern "C" void kernel_launch(void* const* d_in, const int* in_sizes, int n_in,
                              void* d_out, int out_size, void* d_ws, size_t ws_size,
                              hipStream_t stream) {
    const float* fc     = (const float*)d_in[0];
    const float* camera = (const float*)d_in[1];
    const float* rand_u = (const float*)d_in[2];
    const float* W1     = (const float*)d_in[3];
    const float* b1     = (const float*)d_in[4];
    const float* W2     = (const float*)d_in[5];
    const float* b2     = (const float*)d_in[6];
    const int*   a_fb   = (const int*)d_in[7];
    const int*   a_lr   = (const int*)d_in[8];
    const int*   a_jp   = (const int*)d_in[9];
    const int*   a_ss   = (const int*)d_in[10];
    const int*   a_atk  = (const int*)d_in[11];
    float* out = (float*)d_out;

    size_t off = 0;
    auto carve = [&](size_t bytes) {
        void* p = (char*)d_ws + off;
        off += (bytes + 255) & ~(size_t)255;
        return p;
    };
    unsigned short* Xb     = (unsigned short*)carve((size_t)BATCH * KP  * 2);
    unsigned short* W1t    = (unsigned short*)carve((size_t)HID   * KP  * 2);   // [n][k]
    unsigned short* Hb     = (unsigned short*)carve((size_t)BATCH * HID * 2);
    unsigned short* W2t    = (unsigned short*)carve((size_t)NP2   * HID * 2);   // [n][k]
    float*          logits = (float*)         carve((size_t)BATCH * NP2 * 4);
    int*            tgt    = (int*)           carve((size_t)BATCH * 4);

    prep_w1t_kernel<<<dim3(KP / 32, HID / 32), dim3(32, 8), 0, stream>>>(W1, W1t);
    prep_w2t_kernel<<<(NP2 * HID + 255) / 256, 256, 0, stream>>>(W2, W2t);
    conv_fc_kernel<<<((size_t)BATCH * NIN / 4 + 255) / 256, 256, 0, stream>>>(fc, Xb);
    scan_kernel<<<(BATCH + 255) / 256, 256, 0, stream>>>(camera, rand_u,
                                                         a_fb, a_lr, a_jp, a_ss, a_atk,
                                                         Xb, tgt);
    gemm1_kernel<<<dim3(HID / 128, BATCH / 128), 256, 0, stream>>>(Xb, W1t, b1, Hb);
    gemm2_kernel<<<dim3(BATCH / 128), 256, 0, stream>>>(Hb, W2t, b2, logits);
    loss_kernel<<<(BATCH + 255) / 256, 256, 0, stream>>>(logits, tgt, out);
}
